// LocalCausalAttention_45208825758117
// MI455X (gfx1250) — matrix operand order repaired and checked by reading, not compiled, not run on hardware
//
#include <hip/hip_runtime.h>
#include <hip/hip_bf16.h>

#define SEQ      4096
#define DMODEL   1024
#define NHEADS   16
#define DHEAD    64
#define WINDOW   64
#define QKSCALE  0.125f  // 1/sqrt(64)

typedef __attribute__((ext_vector_type(16))) __bf16 v16bf;
typedef __attribute__((ext_vector_type(8)))  float  v8f;
typedef __attribute__((ext_vector_type(4)))  int    i32x4;

union FragBF { unsigned int u[8]; v16bf v; };
static_assert(sizeof(FragBF) == 32, "frag size");

// ---- CDNA5 async global->LDS path (guarded; falls back to VGPR staging) ----
#if defined(__gfx1250__) && __has_builtin(__builtin_amdgcn_global_load_async_to_lds_b128)
#define USE_ASYNC 1
#else
#define USE_ASYNC 0
#endif

#if USE_ASYNC
#define AS1(p) ((__attribute__((address_space(1))) i32x4*)(p))
#define AS3(p) ((__attribute__((address_space(3))) i32x4*)(p))
#if __has_builtin(__builtin_amdgcn_s_wait_asynccnt)
#define WAIT_ASYNC(n) __builtin_amdgcn_s_wait_asynccnt(n)
#else
#define WAIT_ASYNC(n) asm volatile("s_wait_asynccnt %0" :: "i"(n) : "memory")
#endif
#else
#define WAIT_ASYNC(n) do {} while (0)
#endif

__device__ __forceinline__ unsigned short f32_to_bf16(float f) {
    unsigned int x = __builtin_bit_cast(unsigned int, f);
    x += 0x7fffu + ((x >> 16) & 1u);   // round-to-nearest-even
    return (unsigned short)(x >> 16);
}

__device__ __forceinline__ v8f wmma_bf16(const FragBF& a, const FragBF& b, v8f c) {
    return __builtin_amdgcn_wmma_f32_16x16x32_bf16(false, a.v, false, b.v,
                                                   (short)0, c, false, false);
}

// copy 32 contiguous bytes global -> LDS (per-thread)
__device__ __forceinline__ void stage32(const unsigned short* __restrict__ src,
                                        unsigned int* __restrict__ dst) {
#if USE_ASYNC
    __builtin_amdgcn_global_load_async_to_lds_b128(AS1(src), AS3(dst), 0, 0);
    __builtin_amdgcn_global_load_async_to_lds_b128(AS1(src), AS3(dst), 16, 0);
#else
    const uint4* s = (const uint4*)src;
    uint4 p0 = s[0], p1 = s[1];
    dst[0] = p0.x; dst[1] = p0.y; dst[2] = p0.z; dst[3] = p0.w;
    dst[4] = p1.x; dst[5] = p1.y; dst[6] = p1.z; dst[7] = p1.w;
#endif
}

// ---------------------------------------------------------------- fp32 -> bf16
__global__ void cvt_bf16_kernel(const float* __restrict__ in,
                                unsigned short* __restrict__ out, int n) {
    int i = blockIdx.x * blockDim.x + threadIdx.x;
    if (i < n) out[i] = f32_to_bf16(in[i]);
}

// ---------------------------------------------------------------- WMMA GEMM
// C[M,N] = A[M,K](bf16) * W[N,K]^T(bf16) + bias[N]
// Block tile 128x128, K-step 32, double-buffered async LDS staging.
// 256 threads = 8 waves (2x4), each wave computes 64x32 (4x2 WMMA tiles).
__global__ __launch_bounds__(256)
void gemm_bf16_kernel(const unsigned short* __restrict__ A,
                      const unsigned short* __restrict__ W,
                      const float* __restrict__ bias,
                      void* __restrict__ Cout,
                      int M, int N, int K, int out_bf16)
{
    __shared__ unsigned int As[2][128 * 18];  // 128 rows x 32 bf16 (16 dw) + 2 dw pad
    __shared__ unsigned int Bs[2][128 * 18];

    const int bn0  = blockIdx.x * 128;
    const int bm0  = blockIdx.y * 128;
    const int tid  = threadIdx.x;
    const int wave = tid >> 5;
    const int lane = tid & 31;
    const int wm   = wave >> 2;   // 0..1
    const int wn   = wave & 3;    // 0..3
    const int lr   = lane & 15;
    const int lh   = lane >> 4;   // 0/1

    const v8f vzero = {0.f, 0.f, 0.f, 0.f, 0.f, 0.f, 0.f, 0.f};
    v8f acc[4][2];
    for (int i = 0; i < 4; i++)
        for (int j = 0; j < 2; j++) acc[i][j] = vzero;

    const int ldrow  = tid >> 1;   // 0..127, tile row this thread stages
    const int ldhalf = tid & 1;    // which 16-bf16 half of the 32-wide K slab

    const unsigned short* srcA = A + (size_t)(bm0 + ldrow) * K + ldhalf * 16;
    const unsigned short* srcB = W + (size_t)(bn0 + ldrow) * K + ldhalf * 16;
    const int ldsOff = ldrow * 18 + ldhalf * 8;

    const int nIter = K >> 5;

    // prologue: stage tile 0 into buffer 0
    stage32(srcA, &As[0][ldsOff]);
    stage32(srcB, &Bs[0][ldsOff]);

    for (int it = 0; it < nIter; ++it) {
        const int cur = it & 1;
        const bool has_next = (it + 1) < nIter;
        if (has_next) {   // stage tile it+1 into the other buffer (WAR-safe: end barrier of it-1)
            stage32(srcA + (it + 1) * 32, &As[1 - cur][ldsOff]);
            stage32(srcB + (it + 1) * 32, &Bs[1 - cur][ldsOff]);
            WAIT_ASYNC(4);   // current tile's 4 async ops done; next 4 still in flight
        } else {
            WAIT_ASYNC(0);
        }
        __syncthreads();

        FragBF af[4];
        for (int mt = 0; mt < 4; mt++) {
            const unsigned int* arow = &As[cur][(wm * 64 + mt * 16 + lr) * 18];
            const int kb2 = lh * 4;
            for (int v = 0; v < 4; v++) {
                af[mt].u[v]     = arow[kb2 + v];        // K = kbase + 2v, kbase = lh*8
                af[mt].u[4 + v] = arow[8 + kb2 + v];    // K = 16 + kbase + 2v
            }
        }
        FragBF bfv[2];
        for (int nt = 0; nt < 2; nt++) {
            const unsigned int* brow = &Bs[cur][(wn * 32 + nt * 16 + lr) * 18];
            for (int v = 0; v < 4; v++) { bfv[nt].u[v] = brow[lh * 4 + v]; bfv[nt].u[4 + v] = brow[8 + lh * 4 + v]; }  // kbase = lh*8, +16
        }
        for (int mt = 0; mt < 4; mt++)
            for (int nt = 0; nt < 2; nt++)
                acc[mt][nt] = wmma_bf16(af[mt], bfv[nt], acc[mt][nt]);

        __syncthreads();
    }

    // epilogue: C/D layout — lane (lh,lr): N = lr, M = lh*8 + vgpr
    for (int mt = 0; mt < 4; mt++) {
        for (int nt = 0; nt < 2; nt++) {
            const int col   = bn0 + wn * 32 + nt * 16 + lr;
            const float bv  = bias ? bias[col] : 0.0f;
            const int mbase = bm0 + wm * 64 + mt * 16 + lh * 8;
            if (out_bf16) {
                unsigned short* C = (unsigned short*)Cout;
                for (int r = 0; r < 8; r++)
                    C[(size_t)(mbase + r) * N + col] = f32_to_bf16(acc[mt][nt][r] + bv);
            } else {
                float* C = (float*)Cout;
                for (int r = 0; r < 8; r++)
                    C[(size_t)(mbase + r) * N + col] = acc[mt][nt][r] + bv;
            }
        }
    }
}

// ---------------------------------------------------------------- attention
// One block = one head x 64 queries (4 waves x 16 queries).
// qkv: [4096][3072] bf16 (Q at col 0, K at 1024, V at 2048). attn_out: [4096][1024] bf16.
__global__ __launch_bounds__(128)
void attn_kernel(const unsigned short* __restrict__ qkv,
                 unsigned short* __restrict__ attn_out)
{
    __shared__ unsigned int Ksh[128 * 33];   // 128 keys x 64 bf16 (32 dw) + 1 dw pad
    __shared__ unsigned int Vt[64 * 81];     // 64 d x 160 bf16 cols (80 dw) + 1 pad, zero-padded
    __shared__ float        Ssh[4][16 * 80]; // per-wave scores
    __shared__ unsigned int Psh[4][16 * 49]; // per-wave probs, 96 bf16 (48 dw) + 1 pad

    const int h      = blockIdx.y;
    const int qb0    = blockIdx.x * 64;
    const int kstart = qb0 - 63;             // union key window: [kstart, kstart+127]
    const int tid    = threadIdx.x;
    const int w      = tid >> 5;
    const int lane   = tid & 31;
    const int lr     = lane & 15;
    const int lh     = lane >> 4;

    const v8f vzero = {0.f, 0.f, 0.f, 0.f, 0.f, 0.f, 0.f, 0.f};

    // zero Vt (covers padding columns and out-of-range keys)
    for (int i = tid; i < 64 * 81; i += 128) Vt[i] = 0u;
    __syncthreads();

    // stage K (row-major, async) and V (transposed): thread t owns key index t
    {
        const int kk = tid;                  // 0..127
        const int n  = kstart + kk;
        unsigned int* kd = &Ksh[kk * 33];
        if (n >= 0) {
            const unsigned short* ksrc = qkv + (size_t)n * 3072 + 1024 + h * 64;
#if USE_ASYNC
            __builtin_amdgcn_global_load_async_to_lds_b128(AS1(ksrc), AS3(kd),  0, 0);
            __builtin_amdgcn_global_load_async_to_lds_b128(AS1(ksrc), AS3(kd), 16, 0);
            __builtin_amdgcn_global_load_async_to_lds_b128(AS1(ksrc), AS3(kd), 32, 0);
            __builtin_amdgcn_global_load_async_to_lds_b128(AS1(ksrc), AS3(kd), 48, 0);
            __builtin_amdgcn_global_load_async_to_lds_b128(AS1(ksrc), AS3(kd), 64, 0);
            __builtin_amdgcn_global_load_async_to_lds_b128(AS1(ksrc), AS3(kd), 80, 0);
            __builtin_amdgcn_global_load_async_to_lds_b128(AS1(ksrc), AS3(kd), 96, 0);
            __builtin_amdgcn_global_load_async_to_lds_b128(AS1(ksrc), AS3(kd), 112, 0);
#else
            const uint4* src = (const uint4*)ksrc;
            for (int i = 0; i < 8; i++) {
                uint4 p = src[i];
                kd[i * 4 + 0] = p.x; kd[i * 4 + 1] = p.y;
                kd[i * 4 + 2] = p.z; kd[i * 4 + 3] = p.w;
            }
#endif
            const unsigned short* vsrc = qkv + (size_t)n * 3072 + 2048 + h * 64;
            unsigned short* vt16 = (unsigned short*)Vt;
            for (int d = 0; d < 64; d++) vt16[d * 162 + kk] = vsrc[d];
        } else {
            for (int i = 0; i < 32; i++) kd[i] = 0u;
        }
    }
    WAIT_ASYNC(0);
    __syncthreads();

    // per-wave Q A-fragments (16 queries x 64 dhead = 2 K-chunks), from global
    const int q_lane = qb0 + w * 16 + lr;
    const unsigned int* qrow = (const unsigned int*)qkv + (size_t)q_lane * 1536 + h * 32;
    FragBF aQ[2];
    for (int kc = 0; kc < 2; kc++) {
        const int base = kc * 16 + lh * 4;
        for (int v = 0; v < 4; v++) {
            aQ[kc].u[v]     = qrow[base + v];
            aQ[kc].u[4 + v] = qrow[base + 8 + v];
        }
    }

    // scores: 5 key tiles of 16, each = 2 WMMA over dhead
    for (int kt = 0; kt < 5; kt++) {
        v8f acc = vzero;
        for (int kc = 0; kc < 2; kc++) {
            FragBF bK;
            const unsigned int* krow = &Ksh[(w * 16 + kt * 16 + lr) * 33];
            for (int v = 0; v < 4; v++) { bK.u[v] = krow[kc * 16 + lh * 4 + v]; bK.u[4 + v] = krow[kc * 16 + lh * 4 + 8 + v]; }
            acc = wmma_bf16(aQ[kc], bK, acc);
        }
        for (int r = 0; r < 8; r++) {
            const int m   = lh * 8 + r;
            const int key = kstart + w * 16 + kt * 16 + lr;
            const int q   = qb0 + w * 16 + m;
            const bool valid = (key >= 0) && (key <= q) && (q - key < WINDOW);
            Ssh[w][m * 80 + kt * 16 + lr] = valid ? acc[r] * QKSCALE : -__builtin_inff();
        }
    }
    __syncthreads();

    // softmax over 80 keys: lanes 0..15 each own one query row
    if (lane < 16) {
        float* srow = &Ssh[w][lane * 80];
        float mx = -__builtin_inff();
        for (int j = 0; j < 80; j++) mx = fmaxf(mx, srow[j]);
        float sum = 0.f;
        for (int j = 0; j < 80; j++) { float e = __expf(srow[j] - mx); srow[j] = e; sum += e; }
        const float inv = 1.0f / sum;
        unsigned int* prow = &Psh[w][lane * 49];
        for (int j = 0; j < 40; j++) {
            unsigned int lo = f32_to_bf16(srow[2 * j] * inv);
            unsigned int hi = f32_to_bf16(srow[2 * j + 1] * inv);
            prow[j] = lo | (hi << 16);
        }
        for (int j = 40; j < 49; j++) prow[j] = 0u;  // zero-pad keys 80..97
    }
    __syncthreads();

    // P(16x96) @ V(96x64): 4 d-tiles x 3 key-chunks of 32
    for (int nt = 0; nt < 4; nt++) {
        v8f acc = vzero;
        for (int kc = 0; kc < 3; kc++) {
            FragBF aP, bV;
            const unsigned int* prow = &Psh[w][lr * 49];
            const int base = kc * 16 + lh * 4;
            for (int v = 0; v < 4; v++) {
                aP.u[v]     = prow[base + v];
                aP.u[4 + v] = prow[base + 8 + v];
            }
            const unsigned int* vrow = &Vt[(nt * 16 + lr) * 81];
            const int vb = w * 8 + kc * 16 + lh * 4;   // window offset 16w, kbase lh*8, +16
            for (int v = 0; v < 4; v++) { bV.u[v] = vrow[vb + v]; bV.u[4 + v] = vrow[vb + 8 + v]; }
            acc = wmma_bf16(aP, bV, acc);
        }
        for (int r = 0; r < 8; r++) {
            const int m   = lh * 8 + r;
            const int q   = qb0 + w * 16 + m;
            const int col = h * 64 + nt * 16 + lr;
            attn_out[(size_t)q * 1024 + col] = f32_to_bf16(acc[r]);
        }
    }
}

// ---------------------------------------------------------------- launch
extern "C" void kernel_launch(void* const* d_in, const int* in_sizes, int n_in,
                              void* d_out, int out_size, void* d_ws, size_t ws_size,
                              hipStream_t stream)
{
    const float* x     = (const float*)d_in[0];  // [4096,1024]
    const float* w_qkv = (const float*)d_in[1];  // [3072,1024]
    const float* b_qkv = (const float*)d_in[2];  // [3072]
    const float* w_out = (const float*)d_in[3];  // [1024,1024]
    const float* b_out = (const float*)d_in[4];  // [1024]
    float* out = (float*)d_out;                  // [4096,1024] f32

    char* ws = (char*)d_ws;
    unsigned short* x_bf    = (unsigned short*)(ws);                        //  8 MB
    unsigned short* wqkv_bf = (unsigned short*)(ws + ( 8ull << 20));        //  6 MB
    unsigned short* wout_bf = (unsigned short*)(ws + (14ull << 20));        //  2 MB
    unsigned short* qkv_bf  = (unsigned short*)(ws + (16ull << 20));        // 24 MB
    unsigned short* attn_bf = (unsigned short*)(ws + (40ull << 20));        //  8 MB (end 48 MB)

    const int nx = SEQ * DMODEL;          // 4194304
    const int nq = 3 * DMODEL * DMODEL;   // 3145728
    const int no = DMODEL * DMODEL;       // 1048576
    cvt_bf16_kernel<<<(nx + 255) / 256, 256, 0, stream>>>(x,     x_bf,    nx);
    cvt_bf16_kernel<<<(nq + 255) / 256, 256, 0, stream>>>(w_qkv, wqkv_bf, nq);
    cvt_bf16_kernel<<<(no + 255) / 256, 256, 0, stream>>>(w_out, wout_bf, no);

    // QKV projection: [4096,3072] = x @ w_qkv^T + b_qkv  (bf16 out)
    gemm_bf16_kernel<<<dim3(3 * DMODEL / 128, SEQ / 128), 256, 0, stream>>>(
        x_bf, wqkv_bf, b_qkv, (void*)qkv_bf, SEQ, 3 * DMODEL, DMODEL, 1);

    // windowed attention
    attn_kernel<<<dim3(SEQ / 64, NHEADS), 128, 0, stream>>>(qkv_bf, attn_bf);

    // output projection: [4096,1024] = attn @ w_out^T + b_out  (f32 out)
    gemm_bf16_kernel<<<dim3(DMODEL / 128, SEQ / 128), 256, 0, stream>>>(
        attn_bf, wout_bf, b_out, (void*)out, SEQ, DMODEL, DMODEL, 0);
}
